// GSGNN_Sub0_4879082848684
// MI455X (gfx1250) — compile-verified, matrix operand order checked by reading, not run on hardware
//
#include <hip/hip_runtime.h>

typedef __bf16 bf16;
typedef __attribute__((ext_vector_type(16))) __bf16 v16bf;
typedef __attribute__((ext_vector_type(8)))  float    v8f;
typedef __attribute__((ext_vector_type(4)))  unsigned v4u;

#define BN_EPS 1e-5f
#define BM 256
#define BNT 64
#define BK 32

// ---------------- gather descriptor: A rows assembled from up to 8 concat segments ----
struct Seg { const float* p; const int* idx; int ld; int w; };
struct Gather { Seg s[8]; int ns; };

__device__ __forceinline__ unsigned short f2bfu(float f) {
  unsigned u = __float_as_uint(f);
  return (unsigned short)((u + 0x7fffu + ((u >> 16) & 1u)) >> 16);  // RNE
}
__device__ __forceinline__ unsigned fenc(float f) {       // order-preserving float->uint
  unsigned u = __float_as_uint(f);
  return (u & 0x80000000u) ? ~u : (u | 0x80000000u);
}
__device__ __forceinline__ float fdec(unsigned e) {
  return (e & 0x80000000u) ? __uint_as_float(e & 0x7fffffffu) : __uint_as_float(~e);
}

__device__ float gatherA(const Gather& g, int row, int k) {
  #pragma unroll 1
  for (int i = 0; i < g.ns; i++) {
    int w = g.s[i].w;
    if (k < w) {
      const float* p = g.s[i].p;
      if (p == nullptr) return 0.f;                       // zero segment (efcc0)
      int r = g.s[i].idx ? g.s[i].idx[row] : row;
      return p[(long long)r * g.s[i].ld + k];
    }
    k -= w;
  }
  return 0.f;                                             // K padding
}

// ------------- weight conversion: fp32 [K,N] -> bf16 TRANSPOSED [Np,Kp] zero-padded ---
__global__ void k_convert_w(const float* __restrict__ W, bf16* __restrict__ out,
                            int K, int N, int Kp, int Np) {
  long long t = (long long)blockIdx.x * blockDim.x + threadIdx.x;
  if (t >= (long long)Kp * Np) return;
  int n = (int)(t / Kp), k = (int)(t % Kp);
  float v = (k < K && n < N) ? W[(long long)k * N + n] : 0.f;
  out[t] = __builtin_bit_cast(bf16, f2bfu(v));
}
__global__ void k_pad_bias(const float* __restrict__ b, float* __restrict__ out, int N, int Np) {
  int t = blockIdx.x * blockDim.x + threadIdx.x;
  if (t < Np) out[t] = (t < N) ? b[t] : 0.f;
}

// ---------------- bf16 WMMA gather-GEMM: out[M,Np] = act(gather(A)[M,K] @ W + b) ------
// Block tile 256x64, 8 waves stacked in M; each wave: 32 rows x 64 cols, acc[2][4].
// B tile is DMA'd with GLOBAL_LOAD_ASYNC_TO_LDS_B128 (ASYNCcnt), overlapping the
// A-tile gather/convert; fragments then feed 8 v_wmma_f32_16x16x32_bf16 per wave.
__global__ void __launch_bounds__(256)
k_gemm(Gather g, const bf16* __restrict__ Wt, const float* __restrict__ bias,
       float* __restrict__ out, int M, int Kp, int Np, int relu) {
  __shared__ __align__(16) bf16 As[BM][BK];      // 16 KB
  __shared__ __align__(16) bf16 Bs[BNT][BK];     // 4 KB  (Bs[n][k])
  int tid = threadIdx.x;
  int wave = tid >> 5, lane = tid & 31;
  int hf = lane >> 4, lr = lane & 15;
  int m0 = blockIdx.y * BM;
  int n0 = blockIdx.x * BNT;

  // per-thread B-tile DMA source/dest (uniform across K loop except k0)
  int bn = tid >> 2, bkb = (tid & 3) * 8;
  int gn = n0 + bn; if (gn >= Np) gn = Np - 1;             // clamp: junk cols discarded
  const bf16* bsrc0 = Wt + (long long)gn * Kp + bkb;
  unsigned blds = (unsigned)(unsigned long long)&Bs[bn][bkb];  // flat[31:0] == LDS addr

  v8f acc[2][4] = {};
  for (int k0 = 0; k0 < Kp; k0 += BK) {
    // ---- B tile: async DMA 16B/lane, global -> LDS (no VGPR staging) ----
    {
      const bf16* src = bsrc0 + k0;
      asm volatile("global_load_async_to_lds_b128 %0, %1, off"
                   :: "v"(blds), "v"(src) : "memory");
    }
    // ---- A tile: 256 rows x 32 cols; one row per thread, two 16-elem chunks ----
    {
      int row = tid;
      int grow = m0 + row;
      #pragma unroll 1
      for (int c = 0; c < 2; c++) {
        int kb = c * 16;
        unsigned wbuf[8];
        if (grow < M) {
          // resolve segment once per chunk
          int koff = k0 + kb;
          int i = 0;
          #pragma unroll 1
          while (i < g.ns && koff >= g.s[i].w) { koff -= g.s[i].w; i++; }
          if (i < g.ns && g.s[i].p != nullptr && koff + 16 <= g.s[i].w) {
            const Seg sg = g.s[i];
            int r = sg.idx ? sg.idx[grow] : grow;
            const float* p = sg.p + (long long)r * sg.ld + koff;
            __builtin_prefetch(p + BK, 0, 1);              // next K-step (speculative)
            #pragma unroll
            for (int j = 0; j < 8; j++)
              wbuf[j] = (unsigned)f2bfu(p[2 * j]) | ((unsigned)f2bfu(p[2 * j + 1]) << 16);
          } else {                                         // boundary / narrow / zero segs
            #pragma unroll 1
            for (int j = 0; j < 8; j++) {
              float lo = gatherA(g, grow, k0 + kb + 2 * j);
              float hi = gatherA(g, grow, k0 + kb + 2 * j + 1);
              wbuf[j] = (unsigned)f2bfu(lo) | ((unsigned)f2bfu(hi) << 16);
            }
          }
        } else {
          #pragma unroll
          for (int j = 0; j < 8; j++) wbuf[j] = 0u;
        }
        v4u q0 = {wbuf[0], wbuf[1], wbuf[2], wbuf[3]};
        v4u q1 = {wbuf[4], wbuf[5], wbuf[6], wbuf[7]};
        *(v4u*)&As[row][kb]     = q0;
        *(v4u*)&As[row][kb + 8] = q1;
      }
    }
    asm volatile("s_wait_asynccnt 0" ::: "memory");        // B DMA complete
    __syncthreads();
    // ---- fragments + 8 WMMA per wave ----
    union Frag { v16bf v; v4u q[2]; };
    Frag b[4];
    #pragma unroll
    for (int ni = 0; ni < 4; ni++) {
      int bc = ni * 16 + lr;
      b[ni].q[0] = *(const v4u*)&Bs[bc][hf * 8];
      b[ni].q[1] = *(const v4u*)&Bs[bc][16 + hf * 8];
    }
    #pragma unroll
    for (int mi = 0; mi < 2; mi++) {
      Frag a;
      int ar = wave * 32 + mi * 16 + lr;
      a.q[0] = *(const v4u*)&As[ar][hf * 8];
      a.q[1] = *(const v4u*)&As[ar][16 + hf * 8];
      #pragma unroll
      for (int ni = 0; ni < 4; ni++)
        acc[mi][ni] = __builtin_amdgcn_wmma_f32_16x16x32_bf16(
            false, a.v, false, b[ni].v, (short)0, acc[mi][ni], false, false);
    }
    __syncthreads();
  }
  // epilogue: C layout VGPR v -> row = v + 8*hf, col = lr
  #pragma unroll
  for (int mi = 0; mi < 2; mi++)
    #pragma unroll
    for (int ni = 0; ni < 4; ni++) {
      int col = n0 + ni * 16 + lr;
      if (col >= Np) continue;
      float bv = bias[col];
      #pragma unroll
      for (int v = 0; v < 8; v++) {
        int row = m0 + wave * 32 + mi * 16 + v + 8 * hf;
        if (row >= M) continue;
        float x = acc[mi][ni][v] + bv;
        if (relu) x = x > 0.f ? x : 0.f;
        out[(long long)row * Np + col] = x;
      }
    }
}

// ---------------- sigmoid gate: f[e,j] = x[e,1+j] * sigmoid(x[e,0]) ------------------
__global__ void k_gate(const float* __restrict__ x, float* __restrict__ f,
                       int E, int ldx, int n4) {
  long long t = (long long)blockIdx.x * blockDim.x + threadIdx.x;
  if (t >= (long long)E * n4) return;
  int e = (int)(t / n4), j = (int)(t % n4);
  float x0 = x[(long long)e * ldx];
  float k = 1.f / (1.f + expf(-x0));
  f[(long long)e * n4 + j] = x[(long long)e * ldx + 1 + j] * k;
}

// ---------------- segment reductions (sum / max / min / mean by dst) -----------------
__global__ void k_seg_init(float* S, unsigned* MX, unsigned* MN, float* M4,
                           float* deg, int Nn, int n) {
  long long t = (long long)blockIdx.x * blockDim.x + threadIdx.x;
  if (t >= (long long)Nn * n) return;
  S[t] = 0.f; M4[t] = 0.f;
  MX[t] = fenc(-__builtin_inff());
  MN[t] = fenc(__builtin_inff());
  if ((t % n) == 0) deg[t / n] = 0.f;
}
__global__ void k_seg_scatter(const float* __restrict__ f, const int* __restrict__ dst,
                              int E, int n, float* S, unsigned* MX, unsigned* MN,
                              float* M4, float* deg) {
  long long t = (long long)blockIdx.x * blockDim.x + threadIdx.x;
  if (t >= (long long)E * n) return;
  int e = (int)(t / n), j = (int)(t % n);
  int d = dst[e];
  const float* fr = f + (long long)e * 4 * n;
  long long o = (long long)d * n + j;
  atomicAdd(&S[o], fr[j]);
  atomicMax(&MX[o], fenc(fr[n + j]));
  atomicMin(&MN[o], fenc(fr[2 * n + j]));
  atomicAdd(&M4[o], fr[3 * n + j]);
  if (j == 0) atomicAdd(&deg[d], 1.f);
}
__global__ void k_seg_fin(const float* S, const unsigned* MX, const unsigned* MN,
                          const float* M4, const float* deg, float* stats, int Nn, int n) {
  long long t = (long long)blockIdx.x * blockDim.x + threadIdx.x;
  if (t >= (long long)Nn * n) return;
  int node = (int)(t / n), j = (int)(t % n);
  float dg = deg[node];
  bool has = dg > 0.f;
  float* o = stats + (long long)node * 4 * n;
  o[j]         = S[t];
  o[n + j]     = has ? fdec(MX[t]) : 0.f;
  o[2 * n + j] = has ? fdec(MN[t]) : 0.f;
  o[3 * n + j] = M4[t] / fmaxf(dg, 1.f);
}

// ---------------- batchnorm (training-style: batch mean / biased var) ----------------
__global__ void k_bn_stats(const float* __restrict__ x, long long rows, int ld,
                           float* mean, float* var) {
  int j = blockIdx.x;
  float s = 0.f, s2 = 0.f;
  for (long long r = threadIdx.x; r < rows; r += blockDim.x) {
    float v = x[r * ld + j]; s += v; s2 += v * v;
  }
  __shared__ float sh[256], sh2[256];
  sh[threadIdx.x] = s; sh2[threadIdx.x] = s2; __syncthreads();
  for (int o = 128; o > 0; o >>= 1) {
    if ((int)threadIdx.x < o) { sh[threadIdx.x] += sh[threadIdx.x + o]; sh2[threadIdx.x] += sh2[threadIdx.x + o]; }
    __syncthreads();
  }
  if (threadIdx.x == 0) {
    float mu = sh[0] / (float)rows;
    mean[j] = mu;
    var[j] = sh2[0] / (float)rows - mu * mu;
  }
}
__global__ void k_bn_apply(float* x, long long rows, int n, const float* mean,
                           const float* var, const float* g, const float* b) {
  long long t = (long long)blockIdx.x * blockDim.x + threadIdx.x;
  if (t >= rows * n) return;
  int j = (int)(t % n);
  float v = x[t];
  x[t] = g[j] * (v - mean[j]) * rsqrtf(var[j] + BN_EPS) + b[j];
}

// ---------------- misc elementwise ---------------------------------------------------
__global__ void k_make_nf0(const float* nf, float* out, long long n6) {
  long long t = (long long)blockIdx.x * blockDim.x + threadIdx.x;
  if (t >= n6) return;
  int c = (int)(t % 6);
  float v = nf[t];
  if (c == 1 || c == 2) v = 0.f;
  out[t] = v;
}
__global__ void k_bcast(const float* b, float* out, long long tot, int w) {
  long long t = (long long)blockIdx.x * blockDim.x + threadIdx.x;
  if (t < tot) out[t] = b[(int)(t % w)];
}
__global__ void k_col0(const float* x, float* out, int E, int ld) {
  int e = blockIdx.x * blockDim.x + threadIdx.x;
  if (e < E) out[e] = x[(long long)e * ld];
}

// ---------------- host-side helpers --------------------------------------------------
static inline Seg mkseg(const float* p, int ld, int w) { Seg s; s.p = p; s.idx = nullptr; s.ld = ld; s.w = w; return s; }
static inline Gather gmake() { Gather g; g.ns = 0; return g; }
static inline void gpush(Gather& g, Seg s) { g.s[g.ns++] = s; }

extern "C" void kernel_launch(void* const* d_in, const int* in_sizes, int n_in,
                              void* d_out, int out_size, void* d_ws, size_t ws_size,
                              hipStream_t stream) {
  const float* nf    = (const float*)d_in[0];
  const int* cc_src  = (const int*)d_in[3];
  const int* cc_dst  = (const int*)d_in[4];
  const int* ss_src  = (const int*)d_in[5];
  const int* ss_dst  = (const int*)d_in[6];
  const int Nn = in_sizes[0] / 6;
  const int E  = in_sizes[3];
  (void)n_in; (void)out_size; (void)ws_size;

  char* wsp = (char*)d_ws; size_t off = 0;
  auto alloc = [&](size_t bytes) -> void* {
    void* p = wsp + off; off = (off + bytes + 255) & ~(size_t)255; return p;
  };
  float* nf0    = (float*)alloc((size_t)Nn * 6 * 4);
  float* ss_vc0 = (float*)alloc((size_t)Nn * 32 * 4);
  float* ss_ve0 = (float*)alloc((size_t)E * 32 * 4);
  float* vc1 = (float*)alloc((size_t)Nn * 64 * 4);
  float* vc2 = (float*)alloc((size_t)Nn * 128 * 4);
  float* vc3 = (float*)alloc((size_t)Nn * 64 * 4);
  float* vc4 = (float*)alloc((size_t)Nn * 32 * 4);
  float* ve1 = (float*)alloc((size_t)E * 64 * 4);
  float* ve2 = (float*)alloc((size_t)E * 128 * 4);
  float* ve3 = (float*)alloc((size_t)E * 64 * 4);
  float* cc_vc0 = (float*)alloc((size_t)Nn * 32 * 4);
  float* cc_ve0 = (float*)alloc((size_t)E * 32 * 4);
  float* cvc1 = (float*)alloc((size_t)Nn * 64 * 4);
  float* cvc2 = (float*)alloc((size_t)Nn * 128 * 4);
  float* cvc3 = (float*)alloc((size_t)Nn * 64 * 4);
  float* cve1 = (float*)alloc((size_t)E * 64 * 4);
  float* cve2 = (float*)alloc((size_t)E * 128 * 4);
  float* cve3 = (float*)alloc((size_t)E * 64 * 4);
  float* EH = (float*)alloc((size_t)E * 384 * 4);
  float* EO = (float*)alloc((size_t)E * 528 * 4);
  float* EF = (float*)alloc((size_t)E * 512 * 4);
  float* Sb = (float*)alloc((size_t)Nn * 128 * 4);
  unsigned* MXb = (unsigned*)alloc((size_t)Nn * 128 * 4);
  unsigned* MNb = (unsigned*)alloc((size_t)Nn * 128 * 4);
  float* M4b  = (float*)alloc((size_t)Nn * 128 * 4);
  float* degb = (float*)alloc((size_t)Nn * 4);
  float* stats = (float*)alloc((size_t)Nn * 512 * 4);
  bf16* wconv = (bf16*)alloc((size_t)640 * 544 * 2);
  float* bconv = (float*)alloc(544 * 4);
  float* bn_mean = (float*)alloc(128 * 4);
  float* bn_var  = (float*)alloc(128 * 4);
  float* h_nodes = (float*)alloc((size_t)Nn * 64 * 4);
  float* tg  = (float*)alloc((size_t)E * 64 * 4);
  float* th  = (float*)alloc((size_t)E * 32 * 4);
  float* fin = (float*)alloc((size_t)E * 16 * 4);

  auto F = [&](int i) { return (const float*)d_in[i]; };
  auto gwidth = [](const Gather& g) { int w = 0; for (int i = 0; i < g.ns; i++) w += g.s[i].w; return w; };

  auto gemm = [&](const Gather& g, int M, int pW, int pB, int K, int N, float* outp, int relu) {
    int Kp = (K + 31) & ~31, Np = (N + 15) & ~15;
    long long tot = (long long)Kp * Np;
    k_convert_w<<<(unsigned)((tot + 255) / 256), 256, 0, stream>>>(F(pW), wconv, K, N, Kp, Np);
    k_pad_bias<<<(Np + 63) / 64, 64, 0, stream>>>(F(pB), bconv, N, Np);
    dim3 grid((Np + BNT - 1) / BNT, (M + BM - 1) / BM);
    k_gemm<<<grid, 256, 0, stream>>>(g, wconv, bconv, outp, M, Kp, Np, relu);
  };
  auto bn = [&](float* x, long long rows, int nfeat, int pG, int pB) {
    k_bn_stats<<<nfeat, 256, 0, stream>>>(x, rows, nfeat, bn_mean, bn_var);
    long long tot = rows * nfeat;
    k_bn_apply<<<(unsigned)((tot + 255) / 256), 256, 0, stream>>>(x, rows, nfeat, bn_mean, bn_var, F(pG), F(pB));
  };
  auto msgG = [&](const Gather& nfg, const int* src, const int* dst, const Gather& efg) {
    Gather g = gmake();
    for (int i = 0; i < nfg.ns; i++) { Seg s = nfg.s[i]; s.idx = src; gpush(g, s); }
    for (int i = 0; i < nfg.ns; i++) { Seg s = nfg.s[i]; s.idx = dst; gpush(g, s); }
    for (int i = 0; i < efg.ns; i++) gpush(g, efg.s[i]);
    return g;
  };
  auto gmp = [&](int pb, int n, const Gather& msg_g, const Gather& nf_g, const Gather& ef_g,
                 const int* dst, float* vc_out, float* ve_out) {
    int K1 = gwidth(msg_g), N1 = 3 * n, N2 = 4 * n + 1;
    gemm(msg_g, E, pb + 0, pb + 1, K1, N1, EH, 1);
    Gather gh = gmake(); gpush(gh, mkseg(EH, (N1 + 15) & ~15, N1));
    gemm(gh, E, pb + 2, pb + 3, N1, N2, EO, 0);
    int Np2 = (N2 + 15) & ~15;
    long long tg4 = (long long)E * 4 * n;
    k_gate<<<(unsigned)((tg4 + 255) / 256), 256, 0, stream>>>(EO, EF, E, Np2, 4 * n);
    long long tn = (long long)Nn * n;
    k_seg_init<<<(unsigned)((tn + 255) / 256), 256, 0, stream>>>(Sb, MXb, MNb, M4b, degb, Nn, n);
    long long te = (long long)E * n;
    k_seg_scatter<<<(unsigned)((te + 255) / 256), 256, 0, stream>>>(EF, dst, E, n, Sb, MXb, MNb, M4b, degb);
    k_seg_fin<<<(unsigned)((tn + 255) / 256), 256, 0, stream>>>(Sb, MXb, MNb, M4b, degb, stats, Nn, n);
    Gather gr = nf_g; gpush(gr, mkseg(stats, 4 * n, 4 * n));
    gemm(gr, Nn, pb + 4, pb + 5, gwidth(gr), n, vc_out, 0);
    Gather gv = gmake(); gpush(gv, mkseg(EF, 4 * n, 4 * n));
    for (int i = 0; i < ef_g.ns; i++) gpush(gv, ef_g.s[i]);
    gemm(gv, E, pb + 6, pb + 7, gwidth(gv), n, ve_out, 0);
    bn(vc_out, Nn, n, pb + 8, pb + 9);
    bn(ve_out, E, n, pb + 10, pb + 11);
  };

  // ---- nf0 = nf with cols 1,2 zeroed ----
  long long n6 = (long long)Nn * 6;
  k_make_nf0<<<(unsigned)((n6 + 255) / 256), 256, 0, stream>>>(nf, nf0, n6);

  // ---- ss branch ----
  Gather g_nf0 = gmake(); gpush(g_nf0, mkseg(nf0, 6, 6));
  Gather g_efss = gmake(); gpush(g_efss, mkseg((const float*)d_in[2], 2, 2));
  gemm(g_nf0, Nn, 7, 8, 6, 32, ss_vc0, 0);    // fc_ss_nf
  gemm(g_efss, E, 9, 10, 2, 32, ss_ve0, 0);   // fc_ss_ef

  Gather nfA = gmake(); gpush(nfA, mkseg(ss_vc0, 32, 32));
  Gather efA = gmake(); gpush(efA, mkseg(ss_ve0, 32, 32));
  gmp(11, 64, msgG(nfA, ss_src, ss_dst, efA), nfA, efA, ss_dst, vc1, ve1);

  Gather nfB = gmake(); gpush(nfB, mkseg(vc1, 64, 64));
  Gather efB = gmake(); gpush(efB, mkseg(ve1, 64, 64));
  gmp(23, 128, msgG(nfB, ss_src, ss_dst, efB), nfB, efB, ss_dst, vc2, ve2);

  Gather nfC = gmake(); gpush(nfC, mkseg(vc2, 128, 128)); gpush(nfC, mkseg(vc1, 64, 64));
  Gather efC = gmake(); gpush(efC, mkseg(ve2, 128, 128)); gpush(efC, mkseg(ve1, 64, 64));
  gmp(35, 64, msgG(nfC, ss_src, ss_dst, efC), nfC, efC, ss_dst, vc3, ve3);

  // mlp_ss_nf([vc3, ss_vc0]): 96 -> 64 (relu) -> 32
  Gather gm = gmake(); gpush(gm, mkseg(vc3, 64, 64)); gpush(gm, mkseg(ss_vc0, 32, 32));
  gemm(gm, Nn, 47, 48, 96, 64, h_nodes, 1);
  Gather gmh = gmake(); gpush(gmh, mkseg(h_nodes, 64, 64));
  gemm(gmh, Nn, 49, 50, 64, 32, vc4, 0);

  // ---- cc branch ----
  Gather gcc0 = gmake(); gpush(gcc0, mkseg(nf0, 6, 6)); gpush(gcc0, mkseg(vc4, 32, 32));
  gemm(gcc0, Nn, 55, 56, 38, 32, cc_vc0, 0);  // fc_cc_nf
  long long totb = (long long)E * 32;         // fc_cc_ef on zeros -> bias broadcast
  k_bcast<<<(unsigned)((totb + 255) / 256), 256, 0, stream>>>(F(58), cc_ve0, totb, 32);

  Gather nfD = gmake(); gpush(nfD, mkseg(cc_vc0, 32, 32));
  Gather efD = gmake(); gpush(efD, mkseg(cc_ve0, 32, 32));
  gmp(59, 64, msgG(nfD, cc_src, cc_dst, efD), nfD, efD, cc_dst, cvc1, cve1);

  Gather nfE = gmake(); gpush(nfE, mkseg(cvc1, 64, 64));
  Gather efE = gmake(); gpush(efE, mkseg(cve1, 64, 64));
  gmp(71, 128, msgG(nfE, cc_src, cc_dst, efE), nfE, efE, cc_dst, cvc2, cve2);

  Gather nfF = gmake(); gpush(nfF, mkseg(cvc2, 128, 128)); gpush(nfF, mkseg(cvc1, 64, 64));
  Gather efF = gmake(); gpush(efF, mkseg(cve2, 128, 128)); gpush(efF, mkseg(cve1, 64, 64));
  gmp(83, 64, msgG(nfF, cc_src, cc_dst, efF), nfF, efF, cc_dst, cvc3, cve3);

  // ---- to_edge ----
  Gather vcg = gmake(); gpush(vcg, mkseg(cvc3, 64, 64)); gpush(vcg, mkseg(cc_vc0, 32, 32));
  Gather veg = gmake(); gpush(veg, mkseg(cve3, 64, 64)); gpush(veg, mkseg(cc_ve0, 32, 32));
  gpush(veg, mkseg(nullptr, 0, 1));                      // efcc0 zeros
  Gather te = msgG(vcg, cc_src, cc_dst, veg);            // width 289
  gemm(te, E, 95, 96, 289, 128, EH, 1);                  // mlp1 L1
  Gather t1 = gmake(); gpush(t1, mkseg(EH, 128, 128));
  gemm(t1, E, 97, 98, 128, 96, EO, 1);                   // mlp1 L2
  Gather t2 = gmake(); gpush(t2, mkseg(EO, 96, 96));
  gemm(t2, E, 99, 100, 96, 65, EF, 0);                   // mlp1 L3 (Np=80)
  long long tgt = (long long)E * 64;
  k_gate<<<(unsigned)((tgt + 255) / 256), 256, 0, stream>>>(EF, tg, E, 80, 64);
  Gather t3 = gmake(); gpush(t3, mkseg(tg, 64, 64));
  gemm(t3, E, 101, 102, 64, 32, th, 1);                  // mlp2 L1
  Gather t4 = gmake(); gpush(t4, mkseg(th, 32, 32));
  gemm(t4, E, 103, 104, 32, 1, fin, 0);                  // mlp2 L2 (Np=16)
  k_col0<<<(E + 255) / 256, 256, 0, stream>>>(fin, (float*)d_out, E, 16);
}
